// BitLinear_5111011082882
// MI455X (gfx1250) — compile-verified, matrix operand order checked by reading, not run on hardware
//
#include <hip/hip_runtime.h>
#include <hip/hip_bf16.h>

typedef __attribute__((ext_vector_type(16))) _Float16 v16h;
typedef __attribute__((ext_vector_type(8)))  _Float16 v8h;
typedef __attribute__((ext_vector_type(4)))  _Float16 v4h;
typedef __attribute__((ext_vector_type(8)))  float    v8f;

#define K_DIM 1024
#define N_DIM 1024
#define M_DIM 65536
#define BM 128
#define BN 128
#define BK 32
#define NIT (K_DIM / BK)   // 32 k-steps
#define LDSK 40            // padded row stride in halves (80B) -> conflict-free b128 frag loads

// ---------------- scale reduction ----------------
__global__ __launch_bounds__(256) void k_abs_partial(const float* __restrict__ w,
                                                     float* __restrict__ partials) {
    __shared__ float red[256];
    const int t = threadIdx.x;
    const float* p = w + (size_t)blockIdx.x * 1024;
    float s = 0.f;
#pragma unroll
    for (int i = 0; i < 4; ++i) s += fabsf(p[t + i * 256]);
    red[t] = s;
    __syncthreads();
#pragma unroll
    for (int off = 128; off > 0; off >>= 1) {
        if (t < off) red[t] += red[t + off];
        __syncthreads();
    }
    if (t == 0) partials[blockIdx.x] = red[0];
}

__global__ __launch_bounds__(256) void k_scale(const float* __restrict__ partials,
                                               float* __restrict__ scalep) {
    __shared__ float red[256];
    const int t = threadIdx.x;
    float s = partials[t] + partials[t + 256] + partials[t + 512] + partials[t + 768];
    red[t] = s;
    __syncthreads();
#pragma unroll
    for (int off = 128; off > 0; off >>= 1) {
        if (t < off) red[t] += red[t + off];
        __syncthreads();
    }
    if (t == 0) {
        float mean = red[0] * (1.0f / (1024.0f * 1024.0f));
        *scalep = fmaxf(mean, 1e-5f);
    }
}

// ---------------- ternary quantize: W[n,k] f32 -> {-1,0,1} f16 ----------------
__global__ __launch_bounds__(256) void k_quant(const float* __restrict__ w,
                                               const float* __restrict__ scalep,
                                               _Float16* __restrict__ wq) {
    const float inv = 1.0f / (*scalep);
    const int idx = blockIdx.x * 256 + threadIdx.x;   // 262144 float4 groups
    const float4 v = ((const float4*)w)[idx];
    v4h h;
    h.x = (_Float16)fminf(fmaxf(rintf(v.x * inv), -1.f), 1.f);
    h.y = (_Float16)fminf(fmaxf(rintf(v.y * inv), -1.f), 1.f);
    h.z = (_Float16)fminf(fmaxf(rintf(v.z * inv), -1.f), 1.f);
    h.w = (_Float16)fminf(fmaxf(rintf(v.w * inv), -1.f), 1.f);
    *(v4h*)(wq + (size_t)idx * 4) = h;
}

// ---------------- pipelined WMMA GEMM: out = scale * (x @ ternary(W)^T) ----------------
struct Tile {
    float4 a[4];   // 128x32 f32 A slice (4 float4 per thread)
    v8h    b[2];   // 128x32 f16 B slice (2 v8h per thread)
};

__device__ __forceinline__ void load_tile(Tile& t, const float* __restrict__ x,
                                          const _Float16* __restrict__ wq,
                                          int m0, int n0, int k0, int tid) {
#pragma unroll
    for (int i = 0; i < 4; ++i) {
        const int f  = tid + i * 256;          // 0..1023 float4 slots
        const int r  = f >> 3;
        const int kq = (f & 7) * 4;
        t.a[i] = *(const float4*)(x + (size_t)(m0 + r) * K_DIM + k0 + kq);
    }
#pragma unroll
    for (int i = 0; i < 2; ++i) {
        const int f  = tid + i * 256;          // 0..511 v8h slots
        const int r  = f >> 2;
        const int ko = (f & 3) * 8;
        t.b[i] = *(const v8h*)(wq + (size_t)(n0 + r) * K_DIM + k0 + ko);
    }
}

__device__ __forceinline__ void store_tile(const Tile& t, _Float16* sA, _Float16* sB,
                                           int tid) {
#pragma unroll
    for (int i = 0; i < 4; ++i) {
        const int f  = tid + i * 256;
        const int r  = f >> 3;
        const int kq = (f & 7) * 4;
        v4h h;
        h.x = (_Float16)t.a[i].x; h.y = (_Float16)t.a[i].y;
        h.z = (_Float16)t.a[i].z; h.w = (_Float16)t.a[i].w;
        *(v4h*)(&sA[r * LDSK + kq]) = h;
    }
#pragma unroll
    for (int i = 0; i < 2; ++i) {
        const int f  = tid + i * 256;
        const int r  = f >> 2;
        const int ko = (f & 3) * 8;
        *(v8h*)(&sB[r * LDSK + ko]) = t.b[i];
    }
}

__global__ __launch_bounds__(256) void k_gemm(const float* __restrict__ x,
                                              const _Float16* __restrict__ wq,
                                              const float* __restrict__ scalep,
                                              float* __restrict__ out) {
    __shared__ _Float16 sA[2][BM * LDSK];
    __shared__ _Float16 sB[2][BN * LDSK];

    const int tid  = threadIdx.x;
    const int lane = tid & 31;
    const int wave = tid >> 5;            // 0..7
    const int wm   = wave >> 1;           // 0..3 -> 32 rows each
    const int wn   = wave & 1;            // 0..1 -> 64 cols each
    const int half = lane >> 4;           // 0/1
    const int lrow = lane & 15;

    const int m0 = blockIdx.y * BM;
    const int n0 = blockIdx.x * BN;

    v8f acc[2][4] = {};

    // pipeline prologue: tile 0 -> LDS buf0; tile 1 -> regs
    Tile regs;
    load_tile(regs, x, wq, m0, n0, 0, tid);
    store_tile(regs, sA[0], sB[0], tid);
    load_tile(regs, x, wq, m0, n0, BK, tid);
    __syncthreads();

    for (int it = 0; it < NIT; ++it) {
        const int cur = it & 1;
        const int nxt = cur ^ 1;

        // stage tile it+1 into the other LDS buffer (no conflict with reads of cur)
        if (it + 1 < NIT) store_tile(regs, sA[nxt], sB[nxt], tid);
        // issue global loads for tile it+2 -> hidden under this iteration's WMMAs
        if (it + 2 < NIT) load_tile(regs, x, wq, m0, n0, (it + 2) * BK, tid);
        // warm GL2 a few tiles ahead
        if (it + 4 < NIT) {
            const int f = tid;
            const int r = f >> 3;
            __builtin_prefetch(x + (size_t)(m0 + r) * K_DIM + (it + 4) * BK, 0, 1);
        }

        // -- fragments per ISA 16-bit layout + 8 WMMAs from buffer `cur` --
        const _Float16* cA = sA[cur];
        const _Float16* cB = sB[cur];
        v16h afrag[2];
#pragma unroll
        for (int mi = 0; mi < 2; ++mi) {
            const int row = wm * 32 + mi * 16 + lrow;
            v8h lo = *(const v8h*)(&cA[row * LDSK + half * 8]);       // K 0-7 / 8-15
            v8h hi = *(const v8h*)(&cA[row * LDSK + 16 + half * 8]);  // K 16-23 / 24-31
            afrag[mi] = __builtin_shufflevector(lo, hi, 0,1,2,3,4,5,6,7,
                                                        8,9,10,11,12,13,14,15);
        }
#pragma unroll
        for (int ni = 0; ni < 4; ++ni) {
            const int col = wn * 64 + ni * 16 + lrow;
            v8h lo = *(const v8h*)(&cB[col * LDSK + half * 8]);
            v8h hi = *(const v8h*)(&cB[col * LDSK + 16 + half * 8]);
            v16h bfrag = __builtin_shufflevector(lo, hi, 0,1,2,3,4,5,6,7,
                                                         8,9,10,11,12,13,14,15);
#pragma unroll
            for (int mi = 0; mi < 2; ++mi) {
                acc[mi][ni] = __builtin_amdgcn_wmma_f32_16x16x32_f16(
                    /*neg_a=*/false, afrag[mi],
                    /*neg_b=*/false, bfrag,
                    /*c_mod=*/(short)0, acc[mi][ni],
                    /*reuse_a=*/false, /*reuse_b=*/false);
            }
        }
        __syncthreads();   // buf[nxt] stores visible; buf[cur] reads drained
    }

    // -- epilogue: scale in f32, store per C/D layout --
    const float sc = *scalep;
#pragma unroll
    for (int mi = 0; mi < 2; ++mi) {
        const int gr0 = m0 + wm * 32 + mi * 16 + half * 8;   // M = v + 8*half
#pragma unroll
        for (int ni = 0; ni < 4; ++ni) {
            const int gc = n0 + wn * 64 + ni * 16 + lrow;    // N = lane%16
#pragma unroll
            for (int v = 0; v < 8; ++v) {
                out[(size_t)(gr0 + v) * N_DIM + gc] = acc[mi][ni][v] * sc;
            }
        }
    }
}

extern "C" void kernel_launch(void* const* d_in, const int* in_sizes, int n_in,
                              void* d_out, int out_size, void* d_ws, size_t ws_size,
                              hipStream_t stream) {
    const float* x = (const float*)d_in[0];       // [65536, 1024] f32
    const float* w = (const float*)d_in[1];       // [1024, 1024]  f32
    float* out = (float*)d_out;                   // [65536, 1024] f32

    // workspace layout: [0,2MB) ternary W in f16; then 1024 partials; then scale
    _Float16* wq     = (_Float16*)d_ws;
    float* partials  = (float*)((char*)d_ws + (2u << 20));
    float* scalep    = partials + 1024;

    k_abs_partial<<<1024, 256, 0, stream>>>(w, partials);
    k_scale<<<1, 256, 0, stream>>>(partials, scalep);
    k_quant<<<1024, 256, 0, stream>>>(w, scalep, wq);

    dim3 grid(N_DIM / BN, M_DIM / BM);            // n fastest -> L2 reuse of x panel
    k_gemm<<<grid, 256, 0, stream>>>(x, wq, scalep, out);
}
